// SCALEXDecoder_55654186222292
// MI455X (gfx1250) — matrix operand order repaired and checked by reading, not compiled, not run on hardware
//
#include <hip/hip_runtime.h>

#define N_DOMAINS 4
#define LATENT    64
#define HIDDEN    1024
#define OUT_N     16000
#define BATCH     1024
#define EPS_F     1e-5f

#define NT        64            // N columns per workgroup (4 waves x 16)
#define KP        1032          // padded K stride (elements) in LDS
#define SMEM_BYTES (NT * KP * 2)

typedef __attribute__((ext_vector_type(16))) __bf16 v16bf;
typedef __attribute__((ext_vector_type(8)))  float  v8f;

struct alignas(16) U4 { unsigned int x, y, z, w; };
struct alignas(16) F4 { float x, y, z, w; };
union  AB { U4 u[2]; v16bf v; };

__device__ __forceinline__ unsigned short f2bf(float f) {
  union { float f; unsigned int u; } x; x.f = f;
  unsigned int r = x.u + 0x7FFFu + ((x.u >> 16) & 1u);   // round-to-nearest-even
  return (unsigned short)(r >> 16);
}

// -------- h_pre = z @ W1 + b1  (1024x64 @ 64x1024) --------
__global__ void k_gemm1(const float* __restrict__ z, const float* __restrict__ W1,
                        const float* __restrict__ b1, float* __restrict__ h_pre) {
  __shared__ float zs[LATENT];
  const int row = blockIdx.x, tid = threadIdx.x;
  if (tid < LATENT) zs[tid] = z[row * LATENT + tid];
  __syncthreads();
  float acc0 = 0.f, acc1 = 0.f, acc2 = 0.f, acc3 = 0.f;
  for (int k = 0; k < LATENT; ++k) {
    const float zv = zs[k];
    const float* wr = W1 + k * HIDDEN + tid;
    acc0 += zv * wr[0];
    acc1 += zv * wr[256];
    acc2 += zv * wr[512];
    acc3 += zv * wr[768];
  }
  float* hr = h_pre + (size_t)row * HIDDEN + tid;
  hr[0]   = acc0 + b1[tid];
  hr[256] = acc1 + b1[tid + 256];
  hr[512] = acc2 + b1[tid + 512];
  hr[768] = acc3 + b1[tid + 768];
}

// -------- per-domain BN stats -> scale/shift (deterministic fixed-order) --------
__global__ void k_stats(const float* __restrict__ h_pre, const int* __restrict__ dom,
                        const float* __restrict__ gamma, const float* __restrict__ beta,
                        float* __restrict__ scale, float* __restrict__ shift) {
  __shared__ int ds_[BATCH];
  const int tid = threadIdx.x;
  const int j = blockIdx.x * 256 + tid;
  for (int i = tid; i < BATCH; i += 256) ds_[i] = dom[i];
  __syncthreads();
  float s0 = 0, s1 = 0, s2 = 0, s3 = 0, q0 = 0, q1 = 0, q2 = 0, q3 = 0;
  int c0 = 0, c1 = 0, c2 = 0, c3 = 0;
  for (int b = 0; b < BATCH; ++b) {
    const int d = ds_[b];
    const float v = h_pre[(size_t)b * HIDDEN + j];
    const float vv = v * v;
    s0 += (d == 0) ? v : 0.f;  q0 += (d == 0) ? vv : 0.f;  c0 += (d == 0);
    s1 += (d == 1) ? v : 0.f;  q1 += (d == 1) ? vv : 0.f;  c1 += (d == 1);
    s2 += (d == 2) ? v : 0.f;  q2 += (d == 2) ? vv : 0.f;  c2 += (d == 2);
    s3 += (d == 3) ? v : 0.f;  q3 += (d == 3) ? vv : 0.f;  c3 += (d == 3);
  }
  float S[4] = {s0, s1, s2, s3}, Q[4] = {q0, q1, q2, q3};
  int   C[4] = {c0, c1, c2, c3};
  for (int d = 0; d < 4; ++d) {
    const float cf = (float)(C[d] > 0 ? C[d] : 1);
    const float mu = S[d] / cf;
    const float var = Q[d] / cf - mu * mu;
    const float rs = rsqrtf(var + EPS_F);
    const float sc = gamma[d * HIDDEN + j] * rs;
    scale[d * HIDDEN + j] = sc;
    shift[d * HIDDEN + j] = beta[d * HIDDEN + j] - mu * sc;
  }
}

// -------- hb = bf16(relu(h_pre*scale + shift)) --------
__global__ void k_bnrelu(const float* __restrict__ h_pre, const int* __restrict__ dom,
                         const float* __restrict__ scale, const float* __restrict__ shift,
                         unsigned short* __restrict__ hb) {
  const int row = blockIdx.x, tid = threadIdx.x;
  const int d = dom[row];
  for (int jj = 0; jj < 4; ++jj) {
    const int j = tid + jj * 256;
    float v = h_pre[(size_t)row * HIDDEN + j] * scale[d * HIDDEN + j] + shift[d * HIDDEN + j];
    hb[(size_t)row * HIDDEN + j] = f2bf(fmaxf(v, 0.f));
  }
}

// -------- group rows by domain: perm / cnt / ofs --------
__global__ void k_perm(const int* __restrict__ dom, int* __restrict__ perm,
                       int* __restrict__ cntofs) {
  __shared__ int cnt_s[4], cur[4];
  const int tid = threadIdx.x;
  if (tid < 4) cnt_s[tid] = 0;
  __syncthreads();
  for (int i = tid; i < BATCH; i += 256) atomicAdd(&cnt_s[dom[i]], 1);
  __syncthreads();
  if (tid == 0) {
    int o = 0;
    for (int d = 0; d < 4; ++d) {
      cntofs[d] = cnt_s[d];
      cntofs[4 + d] = o;
      cur[d] = o;
      o += cnt_s[d];
    }
  }
  __syncthreads();
  for (int i = tid; i < BATCH; i += 256) {
    const int p = atomicAdd(&cur[dom[i]], 1);
    perm[p] = i;
  }
}

// -------- main routed GEMM: BF16 WMMA, full-K weight slab staged in 132KB LDS --------
// One launch per head (template) -> branch-free epilogue, compile-time W/bias select.
template <int HEAD>
__global__ __launch_bounds__(128) void k_main(
    const unsigned short* __restrict__ hb, const int* __restrict__ perm,
    const int* __restrict__ cntofs,
    const float* __restrict__ Wh, const float* __restrict__ bh,
    float* __restrict__ out) {
  extern __shared__ unsigned short lds[];           // [NT][KP] bf16, N-major
  const int dom   = blockIdx.y;
  const int nbase = blockIdx.x * NT;
  const float* W  = Wh + (size_t)dom * HIDDEN * OUT_N;
  const float* bv = bh + (size_t)dom * OUT_N;
  const int cnt = cntofs[dom];
  const int ofs = cntofs[4 + dom];
  const int tid = threadIdx.x;

  // Stage W[0:1024, nbase:nbase+64] fp32 -> bf16 LDS, transposed to [n][k].
  // 16 float4 groups cover 64 columns; 8 k-rows per iteration; 128 iterations.
  for (int it = 0; it < (HIDDEN * (NT / 4)) / 128; ++it) {
    const int idx = tid + it * 128;
    const int k = idx >> 4;
    const int n = (idx & 15) * 4;
    const F4 f = *reinterpret_cast<const F4*>(W + (size_t)k * OUT_N + nbase + n);
    lds[(n + 0) * KP + k] = f2bf(f.x);
    lds[(n + 1) * KP + k] = f2bf(f.y);
    lds[(n + 2) * KP + k] = f2bf(f.z);
    lds[(n + 3) * KP + k] = f2bf(f.w);
  }
  __syncthreads();

  const int wave = tid >> 5;
  const int lane = tid & 31;
  const int lhi  = lane >> 4;                       // 0 | 1
  const int lo   = lane & 15;
  const int n_loc = wave * 16 + lo;                 // this lane's output column (local)
  const float bias = bv[nbase + n_loc];
  const U4* ldsq = reinterpret_cast<const U4*>(lds) + (size_t)n_loc * (KP / 8);
  const int mtiles = (cnt + 15) >> 4;

  for (int m = 0; m < mtiles; ++m) {
    const int gpa = m * 16 + lo;                    // A: lane holds row (L&15) of tile
    const int rowA = perm[ofs + (gpa < cnt ? gpa : 0)];
    const U4* arow = reinterpret_cast<const U4*>(hb + (size_t)rowA * HIDDEN);
    v8f acc = {};
#pragma unroll 4
    for (int kt = 0; kt < HIDDEN / 32; ++kt) {
      AB a, b;
      // A lanes 0-15: K kt*32+{0..7,16..23}; lanes 16-31: +8
      a.u[0] = arow[kt * 4 + lhi];
      a.u[1] = arow[kt * 4 + lhi + 2];
      // B lanes 0-15: K kt*32+{0..7,16..23}; lanes 16-31: +8
      b.u[0] = ldsq[kt * 4 + lhi];
      b.u[1] = ldsq[kt * 4 + lhi + 2];
      acc = __builtin_amdgcn_wmma_f32_16x16x32_bf16(false, a.v, false, b.v,
                                                    (short)0, acc, false, false);
    }
#pragma unroll
    for (int r = 0; r < 8; ++r) {                   // C/D: VGPR r -> M=r (+8 for hi lanes)
      const int gp = m * 16 + r + lhi * 8;
      if (gp < cnt) {
        const int row = perm[ofs + gp];
        float v = acc[r] + bias;
        if (HEAD == 2) v = __builtin_amdgcn_rcpf(1.0f + __expf(-v));  // fused sigmoid
        out[(size_t)HEAD * BATCH * OUT_N + (size_t)row * OUT_N + nbase + n_loc] = v;
      }
    }
  }
}

// -------- row softmax for heads 0,1 (3-pass, L2-resident rows) --------
__global__ void k_softmax(float* __restrict__ out) {
  __shared__ float red[256];
  const int head = blockIdx.x >> 10;
  const int row  = blockIdx.x & 1023;
  float* p = out + (size_t)head * BATCH * OUT_N + (size_t)row * OUT_N;
  const int tid = threadIdx.x;
  float m = -3.4e38f;
  for (int i = tid; i < OUT_N; i += 256) m = fmaxf(m, p[i]);
  red[tid] = m; __syncthreads();
  for (int s = 128; s > 0; s >>= 1) {
    if (tid < s) red[tid] = fmaxf(red[tid], red[tid + s]);
    __syncthreads();
  }
  const float rmax = red[0]; __syncthreads();
  float sum = 0.f;
  for (int i = tid; i < OUT_N; i += 256) sum += __expf(p[i] - rmax);
  red[tid] = sum; __syncthreads();
  for (int s = 128; s > 0; s >>= 1) {
    if (tid < s) red[tid] += red[tid + s];
    __syncthreads();
  }
  const float inv = 1.0f / red[0]; __syncthreads();
  for (int i = tid; i < OUT_N; i += 256) p[i] = __expf(p[i] - rmax) * inv;
}

extern "C" void kernel_launch(void* const* d_in, const int* in_sizes, int n_in,
                              void* d_out, int out_size, void* d_ws, size_t ws_size,
                              hipStream_t stream) {
  const float* z     = (const float*)d_in[0];
  const int*   dom   = (const int*)d_in[1];
  const float* W1    = (const float*)d_in[2];
  const float* b1    = (const float*)d_in[3];
  const float* gamma = (const float*)d_in[4];
  const float* beta  = (const float*)d_in[5];
  const float* Wm    = (const float*)d_in[6];
  const float* bm    = (const float*)d_in[7];
  const float* Wd    = (const float*)d_in[8];
  const float* bd    = (const float*)d_in[9];
  const float* Wp    = (const float*)d_in[10];
  const float* bp    = (const float*)d_in[11];
  float* out = (float*)d_out;

  char* w = (char*)d_ws;
  float*          h_pre = (float*)w;                               // 4 MB
  unsigned short* hb    = (unsigned short*)(w + (4u << 20));       // 2 MB bf16
  float*          scale = (float*)(w + (6u << 20));                // 16 KB
  float*          shift = scale + N_DOMAINS * HIDDEN;              // 16 KB
  int*            perm  = (int*)(shift + N_DOMAINS * HIDDEN);      // 4 KB
  int*            cntofs = perm + BATCH;                           // 8 ints

  (void)in_sizes; (void)n_in; (void)out_size; (void)ws_size;

  hipFuncSetAttribute(reinterpret_cast<const void*>(k_main<0>),
                      hipFuncAttributeMaxDynamicSharedMemorySize, SMEM_BYTES);
  hipFuncSetAttribute(reinterpret_cast<const void*>(k_main<1>),
                      hipFuncAttributeMaxDynamicSharedMemorySize, SMEM_BYTES);
  hipFuncSetAttribute(reinterpret_cast<const void*>(k_main<2>),
                      hipFuncAttributeMaxDynamicSharedMemorySize, SMEM_BYTES);

  k_gemm1 <<<dim3(BATCH),        dim3(256), 0,          stream>>>(z, W1, b1, h_pre);
  k_stats <<<dim3(4),            dim3(256), 0,          stream>>>(h_pre, dom, gamma, beta, scale, shift);
  k_bnrelu<<<dim3(BATCH),        dim3(256), 0,          stream>>>(h_pre, dom, scale, shift, hb);
  k_perm  <<<dim3(1),            dim3(256), 0,          stream>>>(dom, perm, cntofs);

  dim3 grid(OUT_N / NT, N_DOMAINS);
  k_main<0><<<grid, dim3(128), SMEM_BYTES, stream>>>(hb, perm, cntofs, Wm, bm, out);
  k_main<1><<<grid, dim3(128), SMEM_BYTES, stream>>>(hb, perm, cntofs, Wd, bd, out);
  k_main<2><<<grid, dim3(128), SMEM_BYTES, stream>>>(hb, perm, cntofs, Wp, bp, out);

  k_softmax<<<dim3(2 * BATCH),   dim3(256), 0,          stream>>>(out);
}
